// SelfAttentionBlock_27771258536404
// MI455X (gfx1250) — matrix operand order repaired and checked by reading, not compiled, not run on hardware
//
#include <hip/hip_runtime.h>
#include <hip/hip_bf16.h>
#include <math.h>

typedef __attribute__((ext_vector_type(16))) _Float16 v16h;
typedef __attribute__((ext_vector_type(8)))  _Float16 v8h;
typedef __attribute__((ext_vector_type(8)))  float    v8f;

#define DIM_D 512
#define DIM_F 512
#define DIM_L 2304   // 48*48
#define DIM_N 8
#define EPS_IN 1e-5f

__device__ __forceinline__ v8f wmma_16x16x32(v16h a, v16h b, v8f c) {
    // D = A(16x32 f16) * B(32x16 f16) + C(16x16 f32)
    return __builtin_amdgcn_wmma_f32_16x16x32_f16(false, a, false, b, (short)0, c,
                                                  false, false);
}

__device__ __forceinline__ v8f vzero8() {
    v8f z = {0.f, 0.f, 0.f, 0.f, 0.f, 0.f, 0.f, 0.f};
    return z;
}

// A-fragment (16x32, f16) from a row-major f16 matrix: row = base+lane%16,
// contiguous half-chunks at k + (lane/16)*8 and k + 16 + (lane/16)*8.
__device__ __forceinline__ v16h load_a_f16(const _Float16* __restrict__ p,
                                           size_t ld, int rbase, int kb,
                                           int lhalf, int lrow) {
    const _Float16* ar = p + (size_t)(rbase + lrow) * ld + kb + (lhalf << 3);
    v8h lo = *(const v8h*)ar;
    v8h hi = *(const v8h*)(ar + 16);
    return __builtin_shufflevector(lo, hi, 0, 1, 2, 3, 4, 5, 6, 7,
                                   8, 9, 10, 11, 12, 13, 14, 15);
}

// B-fragment (32x16, f16) where B(k,n) is stored as BT[n][k] row-major:
// lane holds column n = nbase+lane%16, half-chunks at kb+(lane/16)*8 and kb+16+(lane/16)*8.
__device__ __forceinline__ v16h load_b_f16(const _Float16* __restrict__ bt,
                                           size_t ld, int nbase, int kb,
                                           int lhalf, int lrow) {
    const _Float16* br = bt + (size_t)(nbase + lrow) * ld + kb + (lhalf << 3);
    v8h lo = *(const v8h*)br;
    v8h hi = *(const v8h*)(br + 16);
    return __builtin_shufflevector(lo, hi, 0, 1, 2, 3, 4, 5, 6, 7,
                                   8, 9, 10, 11, 12, 13, 14, 15);
}

// ---------------------------------------------------------------------------
// Kernel 0a: one-time f32 -> f16 conversion (weights).
// ---------------------------------------------------------------------------
__global__ void __launch_bounds__(256)
cvt_f32_to_f16_kernel(const float* __restrict__ src, _Float16* __restrict__ dst,
                      int n) {
    const int i = blockIdx.x * 256 + threadIdx.x;
    if (i < n) dst[i] = (_Float16)src[i];
}

// ---------------------------------------------------------------------------
// Kernel 0b: x (N,D,L) f32  ->  Xh (N,L,D) f16 transpose.
// Uses CDNA5 async global->LDS loads (ASYNCcnt) for the staging tile:
// coalesced along l on the load side, coalesced along d on the store side.
// ---------------------------------------------------------------------------
__global__ void __launch_bounds__(256)
transpose_x_kernel(const float* __restrict__ x, _Float16* __restrict__ Xh) {
    __shared__ float sT[32 * 33];   // padded to kill bank conflicts

    const int tid    = threadIdx.x;
    const int ltiles = DIM_L / 32;  // 72
    const int dtiles = DIM_D / 32;  // 16
    int blk          = blockIdx.x;  // N * dtiles * ltiles
    const int batch  = blk / (ltiles * dtiles);
    const int rem    = blk % (ltiles * dtiles);
    const int dbase  = (rem / ltiles) * 32;
    const int lbase  = (rem % ltiles) * 32;

    const float* xb = x + (size_t)batch * DIM_D * DIM_L;

    typedef __attribute__((address_space(3))) float lds_f32;

    const int j  = tid & 31;   // l within tile (contiguous in memory)
    const int i0 = tid >> 5;   // 0..7
#pragma unroll
    for (int it = 0; it < 4; ++it) {
        const int i = i0 + it * 8;   // d within tile
        const float* gp = xb + (size_t)(dbase + i) * DIM_L + lbase + j;
        lds_f32* lp = (lds_f32*)&sT[i * 33 + j];
        asm volatile("global_load_async_to_lds_b32 %0, %1, off"
                     :: "v"(lp), "v"(gp) : "memory");
    }
    asm volatile("s_wait_asynccnt 0x0" ::: "memory");
    __syncthreads();

    // transposed, f16, coalesced along d
    const int lrow = tid >> 3;       // 0..31
    const int d0   = (tid & 7) * 4;  // 0..28
    _Float16* op = Xh + ((size_t)batch * DIM_L + lbase + lrow) * DIM_D + dbase + d0;
#pragma unroll
    for (int k = 0; k < 4; ++k)
        op[k] = (_Float16)sT[(d0 + k) * 33 + lrow];
}

// ---------------------------------------------------------------------------
// Kernel 1: Linear -> tanh -> fused double-InstanceNorm, all-f16 WMMA GEMM.
// One workgroup (4 waves) handles 16 rows (l) x all 512 features.
// ---------------------------------------------------------------------------
__global__ void __launch_bounds__(128)
proj_kernel(const _Float16* __restrict__ Xh, const _Float16* __restrict__ Wh,
            const float* __restrict__ bias, _Float16* __restrict__ out,
            int storeTransposed, float extraScale) {
    __shared__ float sTile[16 * DIM_F];
    __shared__ float sPart[2][16][8];
    __shared__ float sMean[16];
    __shared__ float sScale[16];

    const int tid   = threadIdx.x;
    const int wid   = tid >> 5;
    const int lane  = tid & 31;
    const int lhalf = lane >> 4;
    const int lrow  = lane & 15;

    const int rowTile = blockIdx.x;                 // 0 .. N*L/16-1
    const int batch   = rowTile / (DIM_L / 16);
    const int lbase   = (rowTile % (DIM_L / 16)) * 16;
    const int fw      = wid * 128;

    const _Float16* __restrict__ Xb = Xh + (size_t)batch * DIM_L * DIM_D;

    v8f acc[8];
#pragma unroll
    for (int t = 0; t < 8; ++t) acc[t] = vzero8();

    for (int kb = 0; kb < DIM_D; kb += 32) {
        v16h a = load_a_f16(Xb, DIM_D, lbase, kb, lhalf, lrow);
        __builtin_prefetch(Xb + (size_t)(lbase + lrow) * DIM_D + kb + 64, 0, 1);
#pragma unroll
        for (int t = 0; t < 8; ++t) {
            v16h b = load_b_f16(Wh, DIM_D, fw + t * 16, kb, lhalf, lrow);
            acc[t] = wmma_16x16x32(a, b, acc[t]);
        }
    }

    // bias + tanh -> LDS tile (rows 0..15, cols 0..511)
#pragma unroll
    for (int t = 0; t < 8; ++t) {
        const int col = fw + t * 16 + lrow;
        const float bv = bias[col];
#pragma unroll
        for (int r = 0; r < 8; ++r) {
            const int M = (lhalf << 3) + r;
            sTile[M * DIM_F + col] = tanhf(acc[t][r] + bv);
        }
    }
    __syncthreads();

    // Row statistics: 8 threads per row, 64 cols each.
    const int row = tid >> 3;
    const int seg = tid & 7;
    {
        const float* rp = &sTile[row * DIM_F + seg * 64];
        float s = 0.f, ss = 0.f;
#pragma unroll 8
        for (int j = 0; j < 64; ++j) { float v = rp[j]; s += v; ss += v * v; }
        sPart[0][row][seg] = s;
        sPart[1][row][seg] = ss;
    }
    __syncthreads();
    if (tid < 16) {
        float su = 0.f, sq = 0.f;
#pragma unroll
        for (int j = 0; j < 8; ++j) { su += sPart[0][tid][j]; sq += sPart[1][tid][j]; }
        const float mean = su * (1.0f / DIM_F);
        const float var  = sq * (1.0f / DIM_F) - mean * mean;
        const float r1   = rsqrtf(var + EPS_IN);
        // second InstanceNorm of zero-mean data with var v/(v+eps):
        const float var1 = var / (var + EPS_IN);
        const float r2   = rsqrtf(var1 + EPS_IN);
        sMean[tid]  = mean;
        sScale[tid] = r1 * r2 * extraScale;
    }
    __syncthreads();

    const float mean = sMean[row];
    const float sc   = sScale[row];
    const int   gl   = lbase + row;
    if (!storeTransposed) {
        // out[b][l][f], f16 row-major (contiguous 64 halves per thread)
        _Float16* op = out + ((size_t)batch * DIM_L + gl) * DIM_F + seg * 64;
        const float* vp = &sTile[row * DIM_F + seg * 64];
#pragma unroll 8
        for (int j = 0; j < 64; ++j) op[j] = (_Float16)((vp[j] - mean) * sc);
    } else {
        // Vt[b][f][l] so that P@V B-fragments are contiguous over l
        _Float16* op = out + (size_t)batch * DIM_F * DIM_L + gl;
        for (int j = 0; j < 64; ++j) {
            const int col = seg * 64 + j;
            op[(size_t)col * DIM_L] = (_Float16)((sTile[row * DIM_F + col] - mean) * sc);
        }
    }
}

// ---------------------------------------------------------------------------
// Kernel 2: S = Q @ K^T  (scale folded into Q).  Wave tile: 16(M) x 128(N).
// ---------------------------------------------------------------------------
__global__ void __launch_bounds__(128)
scores_kernel(const _Float16* __restrict__ Qh, const _Float16* __restrict__ Kh,
              float* __restrict__ S, int batch) {
    const int tid = threadIdx.x, wid = tid >> 5, lane = tid & 31;
    const int lhalf = lane >> 4, lrow = lane & 15;
    const int gw    = blockIdx.x * 4 + wid;
    const int mtile = gw / (DIM_L / 128);
    const int ntile = gw % (DIM_L / 128);
    const int mbase = mtile * 16, nbase = ntile * 128;

    const _Float16* Qb = Qh + (size_t)batch * DIM_L * DIM_F;
    const _Float16* Kb = Kh + (size_t)batch * DIM_L * DIM_F;

    v8f acc[8];
#pragma unroll
    for (int t = 0; t < 8; ++t) acc[t] = vzero8();

    for (int kb = 0; kb < DIM_F; kb += 32) {
        v16h a = load_a_f16(Qb, DIM_F, mbase, kb, lhalf, lrow);
        __builtin_prefetch(Qb + (size_t)(mbase + lrow) * DIM_F + kb + 64, 0, 1);
#pragma unroll
        for (int t = 0; t < 8; ++t) {
            v16h b = load_b_f16(Kb, DIM_F, nbase + t * 16, kb, lhalf, lrow);
            acc[t] = wmma_16x16x32(a, b, acc[t]);
        }
    }
#pragma unroll
    for (int t = 0; t < 8; ++t) {
        const int col = nbase + t * 16 + lrow;
#pragma unroll
        for (int r = 0; r < 8; ++r) {
            const int M = (lhalf << 3) + r;
            S[(size_t)(mbase + M) * DIM_L + col] = acc[t][r];
        }
    }
}

// ---------------------------------------------------------------------------
// Kernel 3: row softmax (f32 in -> f16 probabilities).  One block per row.
// ---------------------------------------------------------------------------
__global__ void __launch_bounds__(256)
softmax_kernel(const float* __restrict__ S, _Float16* __restrict__ P) {
    __shared__ float red[256];
    const int row = blockIdx.x;
    const int tid = threadIdx.x;
    const float* sr = S + (size_t)row * DIM_L;

    float v[9];
    float mx = -3.4e38f;
#pragma unroll
    for (int j = 0; j < 9; ++j) { v[j] = sr[tid + 256 * j]; mx = fmaxf(mx, v[j]); }
    red[tid] = mx;
    __syncthreads();
    for (int off = 128; off > 0; off >>= 1) {
        if (tid < off) red[tid] = fmaxf(red[tid], red[tid + off]);
        __syncthreads();
    }
    mx = red[0];
    __syncthreads();

    float sum = 0.f;
#pragma unroll
    for (int j = 0; j < 9; ++j) { v[j] = __expf(v[j] - mx); sum += v[j]; }
    red[tid] = sum;
    __syncthreads();
    for (int off = 128; off > 0; off >>= 1) {
        if (tid < off) red[tid] += red[tid + off];
        __syncthreads();
    }
    const float inv = 1.0f / red[0];

    _Float16* pr = P + (size_t)row * DIM_L;
#pragma unroll
    for (int j = 0; j < 9; ++j) pr[tid + 256 * j] = (_Float16)(v[j] * inv);
}

// ---------------------------------------------------------------------------
// Kernel 4: A = P @ V  using Vt (F x L).  Writes f32 straight into d_out.
// ---------------------------------------------------------------------------
__global__ void __launch_bounds__(128)
attnv_kernel(const _Float16* __restrict__ P, const _Float16* __restrict__ Vt,
             float* __restrict__ out, int batch) {
    const int tid = threadIdx.x, wid = tid >> 5, lane = tid & 31;
    const int lhalf = lane >> 4, lrow = lane & 15;
    const int gw    = blockIdx.x * 4 + wid;
    const int mtile = gw / (DIM_F / 128);
    const int ntile = gw % (DIM_F / 128);
    const int mbase = mtile * 16, nbase = ntile * 128;

    const _Float16* Vb = Vt + (size_t)batch * DIM_F * DIM_L;

    v8f acc[8];
#pragma unroll
    for (int t = 0; t < 8; ++t) acc[t] = vzero8();

    for (int kb = 0; kb < DIM_L; kb += 32) {
        v16h a = load_a_f16(P, DIM_L, mbase, kb, lhalf, lrow);
        __builtin_prefetch(P + (size_t)(mbase + lrow) * DIM_L + kb + 64, 0, 1);
#pragma unroll
        for (int t = 0; t < 8; ++t) {
            v16h b = load_b_f16(Vb, DIM_L, nbase + t * 16, kb, lhalf, lrow);
            acc[t] = wmma_16x16x32(a, b, acc[t]);
        }
    }

    float* ob = out + (size_t)batch * DIM_L * DIM_F;
#pragma unroll
    for (int t = 0; t < 8; ++t) {
        const int col = nbase + t * 16 + lrow;
#pragma unroll
        for (int r = 0; r < 8; ++r) {
            const int M = (lhalf << 3) + r;
            ob[(size_t)(mbase + M) * DIM_F + col] = acc[t][r];
        }
    }
}

// ---------------------------------------------------------------------------
extern "C" void kernel_launch(void* const* d_in, const int* in_sizes, int n_in,
                              void* d_out, int out_size, void* d_ws, size_t ws_size,
                              hipStream_t stream) {
    const float* x  = (const float*)d_in[0];
    const float* Wq = (const float*)d_in[1];
    const float* bq = (const float*)d_in[2];
    const float* Wk = (const float*)d_in[3];
    const float* bk = (const float*)d_in[4];
    const float* Wv = (const float*)d_in[5];
    const float* bv = (const float*)d_in[6];
    float* out = (float*)d_out;

    // Workspace layout (~104 MB)
    const size_t QKV_ELEMS = (size_t)DIM_N * DIM_L * DIM_F;   // per-tensor halves
    const size_t W_ELEMS   = (size_t)DIM_F * DIM_D;
    _Float16* Qh  = (_Float16*)d_ws;
    _Float16* Kh  = Qh + QKV_ELEMS;
    _Float16* Vt  = Kh + QKV_ELEMS;
    float*    S   = (float*)(Vt + QKV_ELEMS);
    _Float16* P   = (_Float16*)(S + (size_t)DIM_L * DIM_L);
    _Float16* Xh  = P + (size_t)DIM_L * DIM_L;
    _Float16* Whq = Xh + (size_t)DIM_N * DIM_L * DIM_D;
    _Float16* Whk = Whq + W_ELEMS;
    _Float16* Whv = Whk + W_ELEMS;

    const dim3 b128(128), b256(256);
    const float qScale = 0.04419417382415922f;               // 1/sqrt(512)

    // one-time precision conversion + layout transpose
    const int nW = (int)W_ELEMS;
    cvt_f32_to_f16_kernel<<<(nW + 255) / 256, b256, 0, stream>>>(Wq, Whq, nW);
    cvt_f32_to_f16_kernel<<<(nW + 255) / 256, b256, 0, stream>>>(Wk, Whk, nW);
    cvt_f32_to_f16_kernel<<<(nW + 255) / 256, b256, 0, stream>>>(Wv, Whv, nW);
    const int trGrid = DIM_N * (DIM_D / 32) * (DIM_L / 32);  // 9216
    transpose_x_kernel<<<trGrid, b256, 0, stream>>>(x, Xh);

    const int projGrid = DIM_N * DIM_L / 16;                 // 1152
    proj_kernel<<<projGrid, b128, 0, stream>>>(Xh, Whq, bq, Qh, 0, qScale);
    proj_kernel<<<projGrid, b128, 0, stream>>>(Xh, Whk, bk, Kh, 0, 1.0f);
    proj_kernel<<<projGrid, b128, 0, stream>>>(Xh, Whv, bv, Vt, 1, 1.0f);

    const int scoresGrid = (DIM_L / 16) * (DIM_L / 128) / 4; // 648
    const int attnvGrid  = (DIM_L / 16) * (DIM_F / 128) / 4; // 144
    for (int n = 0; n < DIM_N; ++n) {
        scores_kernel<<<scoresGrid, b128, 0, stream>>>(Qh, Kh, S, n);
        softmax_kernel<<<DIM_L, b256, 0, stream>>>(S, P);
        attnv_kernel<<<attnvGrid, b128, 0, stream>>>(P, Vt, out, n);
    }
}
